// MultiHeadSelfAttention_52982716563909
// MI455X (gfx1250) — matrix operand order repaired and checked by reading, not compiled, not run on hardware
//
#include <hip/hip_runtime.h>

// MHA: B=4, N=2048, C=1024, H=16, D=64. fp32 in/out, bf16 WMMA compute.
#define B_ 4
#define N_ 2048
#define C_ 1024
#define H_ 16
#define D_ 64
#define SCALE_ 0.125f   // D^-0.5

typedef __bf16 bf16_t;
typedef __attribute__((ext_vector_type(16))) __bf16 v16bf;
typedef __attribute__((ext_vector_type(8)))  __bf16 v8bf;
typedef __attribute__((ext_vector_type(4)))  __bf16 v4bf;
typedef __attribute__((ext_vector_type(8)))  float  v8f;

union V16U { v16bf v; v8bf h[2]; };

__device__ __forceinline__ v8f vzero() {
  v8f v;
#pragma unroll
  for (int i = 0; i < 8; ++i) v[i] = 0.f;
  return v;
}

// A-operand (M=16 x K=32, 16-bit): lane = row (lane&15); elems 0-7 = K grp*8+0..7,
// elems 8-15 = K 16+grp*8+0..7  (ISA 7.12.2 "16-bit A-Matrix 16x32")
__device__ __forceinline__ v16bf load_a16x32(const bf16_t* __restrict__ base, int ld) {
  const int lane = threadIdx.x & 31;
  const bf16_t* p = base + (size_t)(lane & 15) * ld + ((lane >> 4) << 3);
  V16U u;
  u.h[0] = *(const v8bf*)(p);
  u.h[1] = *(const v8bf*)(p + 16);
  return u.v;
}

// B-operand (K=32 x N=16, 16-bit): lane = column n (lane&15); elems 0-7 = K grp*8+0..7,
// elems 8-15 = K 16+grp*8+0..7. Memory view: row of a row-major [N,K] matrix.
__device__ __forceinline__ v16bf load_b32x16(const bf16_t* __restrict__ base, int ld) {
  const int lane = threadIdx.x & 31;
  const bf16_t* p = base + (size_t)(lane & 15) * ld + ((lane >> 4) << 3);
  V16U u;
  u.h[0] = *(const v8bf*)(p);
  u.h[1] = *(const v8bf*)(p + 16);
  return u.v;
}

__device__ __forceinline__ v8f wmma_bf16(v16bf a, v16bf b, v8f c) {
  return __builtin_amdgcn_wmma_f32_16x16x32_bf16(false, a, false, b, (short)0, c,
                                                 false, false);
}

// ---------------- fp32 -> bf16 convert (counts are multiples of 4) ----------------
__global__ void cvt_f32_bf16(const float* __restrict__ in, bf16_t* __restrict__ out,
                             int n4) {
  int i = blockIdx.x * blockDim.x + threadIdx.x;
  if (i < n4) {
    float4 f = ((const float4*)in)[i];
    v4bf o;
    o[0] = (bf16_t)f.x; o[1] = (bf16_t)f.y; o[2] = (bf16_t)f.z; o[3] = (bf16_t)f.w;
    ((v4bf*)out)[i] = o;
  }
}

// ---------------- QKV GEMM: [8192,1024] x [3072,1024]^T, scatter to Q/K/Vt ----------
// Workgroup tile 128x128 (8 waves as 2m x 4n); wave tile 64x32 = 4x2 WMMA tiles.
// A frags reused x2, B frags reused x4 -> ~1.5 b128 loads per WMMA.
__global__ __launch_bounds__(256) void qkv_gemm(
    const bf16_t* __restrict__ xb, const bf16_t* __restrict__ wq,
    const float* __restrict__ bqkv,
    bf16_t* __restrict__ Q, bf16_t* __restrict__ K, bf16_t* __restrict__ Vt) {
  const int wv   = threadIdx.x >> 5;
  const int lane = threadIdx.x & 31;
  const int grp  = lane >> 4;
  const int NB   = (3 * C_) / 128;             // 24 n-blocks
  const int mblk = blockIdx.x / NB;
  const int nblk = blockIdx.x % NB;
  const int wm   = wv & 1;                     // 2 m-waves
  const int wn   = wv >> 1;                    // 4 n-waves
  const int m0   = mblk * 128 + wm * 64;       // wave M origin (4 tiles)
  const int n0   = nblk * 128 + wn * 32;       // wave N origin (2 tiles)

  const bf16_t* Abase = xb + (size_t)m0 * C_;
  const bf16_t* Bbase = wq + (size_t)n0 * C_;

  v8f acc[4][2];
#pragma unroll
  for (int i = 0; i < 4; ++i)
#pragma unroll
    for (int j = 0; j < 2; ++j) acc[i][j] = vzero();

#pragma unroll 2
  for (int kk = 0; kk < C_; kk += 32) {
    v16bf af[4], bfr[2];
#pragma unroll
    for (int i = 0; i < 4; ++i) af[i] = load_a16x32(Abase + (size_t)(i * 16) * C_ + kk, C_);
#pragma unroll
    for (int j = 0; j < 2; ++j) bfr[j] = load_b32x16(Bbase + (size_t)(j * 16) * C_ + kk, C_);
#pragma unroll
    for (int i = 0; i < 4; ++i)
#pragma unroll
      for (int j = 0; j < 2; ++j) acc[i][j] = wmma_bf16(af[i], bfr[j], acc[i][j]);
  }

#pragma unroll
  for (int j = 0; j < 2; ++j) {
    const int col  = n0 + j * 16 + (lane & 15);  // 0..3071
    const float bias = bqkv[col];
    const int s = col / C_;                      // 0:q 1:k 2:v
    const int r = col % C_;
    const int h = r / D_;
    const int d = r % D_;
#pragma unroll
    for (int i = 0; i < 4; ++i)
#pragma unroll
      for (int e = 0; e < 8; ++e) {
        const int trow = m0 + i * 16 + e + 8 * grp;  // global token index
        const int b    = trow / N_;
        const int n    = trow % N_;
        const float v  = acc[i][j][e] + bias;
        const size_t bh = (size_t)b * H_ + h;
        if (s == 0)      Q [(bh * N_ + n) * D_ + d] = (bf16_t)(v * SCALE_);
        else if (s == 1) K [(bh * N_ + n) * D_ + d] = (bf16_t)v;
        else             Vt[(bh * D_ + d) * N_ + n] = (bf16_t)v;   // V transposed
      }
  }
}

// ---------------- Flash attention: wave = 32 queries (2 q-tiles), key blocks of 64 ---
// K and V fragments are loaded once per key block and reused across both q-tiles.
__global__ __launch_bounds__(128) void flash_attn(
    const bf16_t* __restrict__ Q, const bf16_t* __restrict__ K,
    const bf16_t* __restrict__ Vt, const int* __restrict__ mask,
    bf16_t* __restrict__ Aout) {
  __shared__ bf16_t lds_p[4][32][64];          // per-wave P tiles (C/D -> A relayout)
  const int wv   = threadIdx.x >> 5;
  const int lane = threadIdx.x & 31;
  const int grp  = lane >> 4;
  const int QB   = N_ / 128;                   // 16 q-blocks per (b,h)
  const int qb   = blockIdx.x % QB;
  const int bh   = blockIdx.x / QB;
  const int b    = bh / H_;
  const int h    = bh % H_;
  const int q0   = qb * 128 + wv * 32;         // wave query origin (2 tiles)

  v16bf qa[2][2];
#pragma unroll
  for (int qt = 0; qt < 2; ++qt) {
    const bf16_t* Qb = Q + ((size_t)bh * N_ + q0 + qt * 16) * D_;
    qa[qt][0] = load_a16x32(Qb, D_);
    qa[qt][1] = load_a16x32(Qb + 32, D_);
  }

  float row_m[2][8], row_l[2][8];
  v8f O[2][4];
#pragma unroll
  for (int qt = 0; qt < 2; ++qt) {
#pragma unroll
    for (int i = 0; i < 8; ++i) { row_m[qt][i] = -3.0e38f; row_l[qt][i] = 0.f; }
#pragma unroll
    for (int t = 0; t < 4; ++t) O[qt][t] = vzero();
  }

  const int* mrow = mask + (size_t)b * N_;

  for (int j0 = 0; j0 < N_; j0 += 64) {
    // prefetch next key block (global_prefetch_b8), spread over rows by lane
    if (j0 + 64 < N_) {
      const bf16_t* Kn = K + ((size_t)bh * N_ + j0 + 64 + lane) * D_;
      __builtin_prefetch(Kn, 0, 1);
      __builtin_prefetch(Kn + (size_t)32 * D_, 0, 1);
      __builtin_prefetch(Vt + ((size_t)bh * D_ + lane) * N_ + j0 + 64, 0, 1);
      __builtin_prefetch(Vt + ((size_t)bh * D_ + 32 + lane) * N_ + j0 + 64, 0, 1);
    }
    // S = (Q*scale) K^T : 16 WMMAs; K frags shared across both q-tiles
    v8f S[2][4];
#pragma unroll
    for (int t = 0; t < 4; ++t) {
      const bf16_t* Kb = K + ((size_t)bh * N_ + j0 + t * 16) * D_;
      const v16bf kf0 = load_b32x16(Kb, D_);
      const v16bf kf1 = load_b32x16(Kb + 32, D_);
#pragma unroll
      for (int qt = 0; qt < 2; ++qt) {
        v8f s = vzero();
        s = wmma_bf16(qa[qt][0], kf0, s);
        s = wmma_bf16(qa[qt][1], kf1, s);
        S[qt][t] = s;
      }
    }
    // key mask (columns live in lanes)
#pragma unroll
    for (int t = 0; t < 4; ++t)
      if (mrow[j0 + t * 16 + (lane & 15)] == 0) {
#pragma unroll
        for (int qt = 0; qt < 2; ++qt)
#pragma unroll
          for (int i = 0; i < 8; ++i) S[qt][t][i] = -1.0e30f;
      }
    // online softmax per q-tile; P written to LDS for A-operand relayout
#pragma unroll
    for (int qt = 0; qt < 2; ++qt) {
      float mnew[8], scv[8];
#pragma unroll
      for (int i = 0; i < 8; ++i) {
        float mx = fmaxf(fmaxf(S[qt][0][i], S[qt][1][i]),
                         fmaxf(S[qt][2][i], S[qt][3][i]));
#pragma unroll
        for (int off = 1; off < 16; off <<= 1) mx = fmaxf(mx, __shfl_xor(mx, off, 16));
        mnew[i] = fmaxf(row_m[qt][i], mx);
        scv[i]  = __expf(row_m[qt][i] - mnew[i]);
        row_m[qt][i] = mnew[i];
      }
      float lsum[8] = {0, 0, 0, 0, 0, 0, 0, 0};
#pragma unroll
      for (int t = 0; t < 4; ++t)
#pragma unroll
        for (int i = 0; i < 8; ++i) {
          const float p = __expf(S[qt][t][i] - mnew[i]);
          lsum[i] += p;
          lds_p[wv][qt * 16 + i + 8 * grp][t * 16 + (lane & 15)] = (bf16_t)p;
        }
#pragma unroll
      for (int i = 0; i < 8; ++i) {
        float s = lsum[i];
#pragma unroll
        for (int off = 1; off < 16; off <<= 1) s += __shfl_xor(s, off, 16);
        row_l[qt][i] = row_l[qt][i] * scv[i] + s;
      }
#pragma unroll
      for (int t = 0; t < 4; ++t)
#pragma unroll
        for (int i = 0; i < 8; ++i) O[qt][t][i] *= scv[i];
    }
    // re-read P in A-operand layout (wave-local, in-order LDS pipe)
    V16U pa[2][2];
#pragma unroll
    for (int qt = 0; qt < 2; ++qt) {
      const bf16_t* pr = &lds_p[wv][qt * 16 + (lane & 15)][grp * 8];
      pa[qt][0].h[0] = *(const v8bf*)(pr);
      pa[qt][0].h[1] = *(const v8bf*)(pr + 16);
      pa[qt][1].h[0] = *(const v8bf*)(pr + 32);
      pa[qt][1].h[1] = *(const v8bf*)(pr + 48);
    }
    // O += P V : V frags shared across both q-tiles
#pragma unroll
    for (int dt = 0; dt < 4; ++dt) {
      const bf16_t* Vb = Vt + ((size_t)bh * D_ + dt * 16) * N_ + j0;
      const v16bf vf0 = load_b32x16(Vb, N_);
      const v16bf vf1 = load_b32x16(Vb + 32, N_);
#pragma unroll
      for (int qt = 0; qt < 2; ++qt) {
        O[qt][dt] = wmma_bf16(pa[qt][0].v, vf0, O[qt][dt]);
        O[qt][dt] = wmma_bf16(pa[qt][1].v, vf1, O[qt][dt]);
      }
    }
  }
  // normalize + store attn-out as bf16 [B,N,C] for the projection GEMM
#pragma unroll
  for (int qt = 0; qt < 2; ++qt)
#pragma unroll
    for (int dt = 0; dt < 4; ++dt)
#pragma unroll
      for (int i = 0; i < 8; ++i) {
        const int rq = q0 + qt * 16 + i + 8 * grp;
        const float o = O[qt][dt][i] / row_l[qt][i];
        Aout[((size_t)b * N_ + rq) * C_ + h * D_ + dt * 16 + (lane & 15)] = (bf16_t)o;
      }
}

// ---------------- Output projection: [8192,1024] x [1024,1024]^T + bias (fp32) ------
__global__ __launch_bounds__(256) void proj_gemm(
    const bf16_t* __restrict__ Ab, const bf16_t* __restrict__ wp,
    const float* __restrict__ bp, float* __restrict__ out) {
  const int wv   = threadIdx.x >> 5;
  const int lane = threadIdx.x & 31;
  const int grp  = lane >> 4;
  const int NB   = C_ / 128;                   // 8 n-blocks
  const int mblk = blockIdx.x / NB;
  const int nblk = blockIdx.x % NB;
  const int wm   = wv & 1;
  const int wn   = wv >> 1;
  const int m0   = mblk * 128 + wm * 64;
  const int n0   = nblk * 128 + wn * 32;

  const bf16_t* Abase = Ab + (size_t)m0 * C_;
  const bf16_t* Bbase = wp + (size_t)n0 * C_;

  v8f acc[4][2];
#pragma unroll
  for (int i = 0; i < 4; ++i)
#pragma unroll
    for (int j = 0; j < 2; ++j) acc[i][j] = vzero();

#pragma unroll 2
  for (int kk = 0; kk < C_; kk += 32) {
    v16bf af[4], bfr[2];
#pragma unroll
    for (int i = 0; i < 4; ++i) af[i] = load_a16x32(Abase + (size_t)(i * 16) * C_ + kk, C_);
#pragma unroll
    for (int j = 0; j < 2; ++j) bfr[j] = load_b32x16(Bbase + (size_t)(j * 16) * C_ + kk, C_);
#pragma unroll
    for (int i = 0; i < 4; ++i)
#pragma unroll
      for (int j = 0; j < 2; ++j) acc[i][j] = wmma_bf16(af[i], bfr[j], acc[i][j]);
  }

#pragma unroll
  for (int j = 0; j < 2; ++j) {
    const int col = n0 + j * 16 + (lane & 15);
    const float bias = bp[col];
#pragma unroll
    for (int i = 0; i < 4; ++i)
#pragma unroll
      for (int e = 0; e < 8; ++e) {
        const int trow = m0 + i * 16 + e + 8 * grp;
        out[(size_t)trow * C_ + col] = acc[i][j][e] + bias;
      }
  }
}

extern "C" void kernel_launch(void* const* d_in, const int* in_sizes, int n_in,
                              void* d_out, int out_size, void* d_ws, size_t ws_size,
                              hipStream_t stream) {
  (void)in_sizes; (void)n_in; (void)out_size; (void)ws_size;
  const float* x      = (const float*)d_in[0];
  const int*   mask   = (const int*)  d_in[1];
  const float* w_qkv  = (const float*)d_in[2];
  const float* b_qkv  = (const float*)d_in[3];
  const float* w_proj = (const float*)d_in[4];
  const float* b_proj = (const float*)d_in[5];
  float* out = (float*)d_out;

  // workspace carve-out (bf16 buffers), ~76 MB total
  size_t off = 0;
  auto carve = [&](size_t elems) -> bf16_t* {
    bf16_t* p = (bf16_t*)((char*)d_ws + off);
    off += (elems * sizeof(bf16_t) + 255) & ~(size_t)255;
    return p;
  };
  const size_t nx = (size_t)B_ * N_ * C_;      // 8.39M
  bf16_t* xb     = carve(nx);
  bf16_t* wqkvb  = carve((size_t)3 * C_ * C_);
  bf16_t* wprojb = carve((size_t)C_ * C_);
  bf16_t* Q      = carve(nx);
  bf16_t* Kb     = carve(nx);
  bf16_t* Vt     = carve(nx);
  bf16_t* Ao     = xb;                         // xb dead after qkv_gemm: alias

  auto cvt = [&](const float* src, bf16_t* dst, size_t n) {
    const int thr = 256;
    const int n4 = (int)(n / 4);
    cvt_f32_bf16<<<(n4 + thr - 1) / thr, thr, 0, stream>>>(src, dst, n4);
  };
  cvt(x, xb, nx);
  cvt(w_qkv, wqkvb, (size_t)3 * C_ * C_);
  cvt(w_proj, wprojb, (size_t)C_ * C_);

  // qkv: 64 m-blocks x 24 n-blocks; proj: 64 x 8; flash: 64 bh x 16 q-blocks
  qkv_gemm <<<64 * 24, 256, 0, stream>>>(xb, wqkvb, b_qkv, Q, Kb, Vt);
  flash_attn<<<64 * 16, 128, 0, stream>>>(Q, Kb, Vt, mask, Ao);
  proj_gemm<<<64 * 8, 256, 0, stream>>>(Ao, wprojb, b_proj, out);
}
